// QuantizedLinear_9371618640414
// MI455X (gfx1250) — matrix operand order repaired and checked by reading, not compiled, not run on hardware
//
#include <hip/hip_runtime.h>
#include <hip/hip_bf16.h>

typedef __bf16 bf16_t;
typedef bf16_t v16bf  __attribute__((ext_vector_type(16)));
typedef bf16_t bf16x8 __attribute__((ext_vector_type(8)));
typedef float  v8f    __attribute__((ext_vector_type(8)));

#define K_DIM 4096
#define N_DIM 11008
#define M_DIM 4096
#define BM 128
#define BN 128
#define BK 32
#define KITERS (K_DIM / BK) /* 128 */

union V16 { v16bf v; bf16x8 h[2]; };

__global__ __launch_bounds__(256)
void w4a16_wmma_kernel(const float* __restrict__ x,
                       const int*   __restrict__ wp,
                       const float* __restrict__ wscale,
                       const float* __restrict__ bias,
                       float*       __restrict__ out)
{
    // double-buffered bf16 staging tiles, row-major [row][k], 64B row stride
    __shared__ alignas(16) bf16_t Xs[2][BM * BK];  // 2 x 8KB
    __shared__ alignas(16) bf16_t Ws[2][BN * BK];  // 2 x 8KB

    const int tid    = threadIdx.x;
    const int lane   = tid & 31;
    const int wave   = tid >> 5;        // 0..7
    const int wm     = wave >> 2;       // 0..1  (M direction)
    const int wn     = wave & 3;        // 0..3  (N direction)
    const int laneLo = lane & 15;
    const int hiSel  = (lane >> 4) & 1; // 0: lanes 0-15, 1: lanes 16-31

    const int nBase = blockIdx.x * BN;
    const int mBase = blockIdx.y * BM;

    // ---------------- cooperative loader mapping ----------------
    // 2 threads per tile row; each covers 16 k-elements (one 32B LDS chunk).
    const int lr = tid >> 1;   // 0..127: row within tile
    const int lh = tid & 1;    // which 16-wide k half

    const float* xg = x  + (size_t)(mBase + lr) * K_DIM + lh * 16;
    const int*   wg = wp + (size_t)(nBase + lr) * (K_DIM / 2) + lh * 8;
    const float  scale = wscale[nBase + lr];

    const int ldsOff = lr * BK + lh * 16;

    float4 xr[4];
    int4   wr[2];

    // ---- load tile 0 into registers ----
    {
        const float4* p = (const float4*)xg;
        xr[0] = p[0]; xr[1] = p[1]; xr[2] = p[2]; xr[3] = p[3];
        const int4* q = (const int4*)wg;
        wr[0] = q[0]; wr[1] = q[1];
    }

    v8f acc[4][2];
#pragma unroll
    for (int i = 0; i < 4; ++i)
#pragma unroll
        for (int j = 0; j < 2; ++j)
            acc[i][j] = (v8f){};

    // ---- stage tile 0 into LDS buffer 0 ----
    {
        const float* xf = (const float*)xr;
        bf16x8 t0, t1;
#pragma unroll
        for (int e = 0; e < 8; ++e) {
            t0[e] = (bf16_t)xf[e];
            t1[e] = (bf16_t)xf[8 + e];
        }
        bf16x8* xd = (bf16x8*)&Xs[0][ldsOff];
        xd[0] = t0; xd[1] = t1;

        const int* wi = (const int*)wr;
        bf16x8 u0, u1;
#pragma unroll
        for (int q = 0; q < 4; ++q) {
            int b  = wi[q] & 0xFF;
            int lo = (b << 28) >> 28;
            int hi = (b << 24) >> 28;
            u0[2 * q]     = (bf16_t)((float)lo * scale);
            u0[2 * q + 1] = (bf16_t)((float)hi * scale);
        }
#pragma unroll
        for (int q = 0; q < 4; ++q) {
            int b  = wi[4 + q] & 0xFF;
            int lo = (b << 28) >> 28;
            int hi = (b << 24) >> 28;
            u1[2 * q]     = (bf16_t)((float)lo * scale);
            u1[2 * q + 1] = (bf16_t)((float)hi * scale);
        }
        bf16x8* wd = (bf16x8*)&Ws[0][ldsOff];
        wd[0] = u0; wd[1] = u1;
    }

    // ---- main loop: ONE barrier per stage (ping-pong LDS) ----
    for (int kt = 0; kt < KITERS; ++kt) {
        __syncthreads();   // buf[kt&1] stores visible; buf[(kt+1)&1] readers done

        const int cur = kt & 1;
        const int nxt = cur ^ 1;
        const bool more = (kt + 1 < KITERS);

        // issue next tile's global loads (overlap with WMMAs below)
        if (more) {
            const float4* p = (const float4*)(xg + (size_t)(kt + 1) * BK);
            xr[0] = p[0]; xr[1] = p[1]; xr[2] = p[2]; xr[3] = p[3];
            const int4* q = (const int4*)(wg + (kt + 1) * (BK / 2));
            wr[0] = q[0]; wr[1] = q[1];
        }

        // ---- B fragments: col = n, 16 contiguous k per lane-half ----
        V16 bfr[2];
#pragma unroll
        for (int j = 0; j < 2; ++j) {
            const int c = wn * 32 + j * 16 + laneLo;
            const bf16x8* p = (const bf16x8*)&Ws[cur][c * BK + hiSel * 16];
            bfr[j].h[0] = p[0];
            bfr[j].h[1] = p[1];
        }

        // ---- A fragments + 8 WMMAs ----
#pragma unroll
        for (int i = 0; i < 4; ++i) {
            const int r = wm * 64 + i * 16 + laneLo;
            V16 afr;
            afr.h[0] = *(const bf16x8*)&Xs[cur][r * BK + hiSel * 16];
            afr.h[1] = *(const bf16x8*)&Xs[cur][r * BK + hiSel * 16 + 8];
#pragma unroll
            for (int j = 0; j < 2; ++j) {
                acc[i][j] = __builtin_amdgcn_wmma_f32_16x16x32_bf16(
                    false, afr.v, false, bfr[j].v,
                    (short)0, acc[i][j], false, false);
            }
        }

        // ---- stage next tile into the other LDS buffer (no barrier needed:
        //      readers of buf[nxt] finished before the barrier above) ----
        if (more) {
            const float* xf = (const float*)xr;
            bf16x8 t0, t1;
#pragma unroll
            for (int e = 0; e < 8; ++e) {
                t0[e] = (bf16_t)xf[e];
                t1[e] = (bf16_t)xf[8 + e];
            }
            bf16x8* xd = (bf16x8*)&Xs[nxt][ldsOff];
            xd[0] = t0; xd[1] = t1;

            const int* wi = (const int*)wr;
            bf16x8 u0, u1;
#pragma unroll
            for (int q = 0; q < 4; ++q) {
                int b  = wi[q] & 0xFF;
                int lo = (b << 28) >> 28;
                int hi = (b << 24) >> 28;
                u0[2 * q]     = (bf16_t)((float)lo * scale);
                u0[2 * q + 1] = (bf16_t)((float)hi * scale);
            }
#pragma unroll
            for (int q = 0; q < 4; ++q) {
                int b  = wi[4 + q] & 0xFF;
                int lo = (b << 28) >> 28;
                int hi = (b << 24) >> 28;
                u1[2 * q]     = (bf16_t)((float)lo * scale);
                u1[2 * q + 1] = (bf16_t)((float)hi * scale);
            }
            bf16x8* wd = (bf16x8*)&Ws[nxt][ldsOff];
            wd[0] = u0; wd[1] = u1;
        }
    }

    // ---------------- epilogue: + bias, non-temporal fp32 stores ----------------
#pragma unroll
    for (int j = 0; j < 2; ++j) {
        const int col = nBase + wn * 32 + j * 16 + laneLo;
        const float bj = bias[col];
#pragma unroll
        for (int i = 0; i < 4; ++i) {
            const int rowBase = mBase + wm * 64 + i * 16 + hiSel * 8;
#pragma unroll
            for (int r = 0; r < 8; ++r) {
                float v = acc[i][j][r] + bj;
                __builtin_nontemporal_store(v, &out[(size_t)(rowBase + r) * N_DIM + col]);
            }
        }
    }
}

extern "C" void kernel_launch(void* const* d_in, const int* in_sizes, int n_in,
                              void* d_out, int out_size, void* d_ws, size_t ws_size,
                              hipStream_t stream) {
    const float* x      = (const float*)d_in[0];
    const int*   wp     = (const int*)d_in[1];
    const float* wscale = (const float*)d_in[2];
    const float* bias   = (const float*)d_in[3];
    float*       out    = (float*)d_out;

    dim3 grid(N_DIM / BN, M_DIM / BM); // (86, 32)
    w4a16_wmma_kernel<<<grid, 256, 0, stream>>>(x, wp, wscale, bias, out);
}
